// ChromaticResonance_37203006718584
// MI455X (gfx1250) — compile-verified, matrix operand order checked by reading, not run on hardware
//
#include <hip/hip_runtime.h>

// ---------------------------------------------------------------------------
// ChromaticResonance fused kernel for gfx1250 (MI455X), wave32 + WMMA bf16.
//
// Per depth: chamber[B,512] (complex f32) x 5 real 512x512 matrices, harmonic
// mixing, tanh, damping. Rows independent -> 32 rows per workgroup, chamber
// resident in LDS (double-buffered bf16) across all 7 depths.
//
// Harmonics are separable per matrix (acc = sum_j f_j(hw_j)), so the GEMMs
// run as two K-sweeps folded into a running sum, keeping peak VGPR use <256
// (no s_set_vgpr_msb overhead, 2 waves/SIMD guaranteed).
// ---------------------------------------------------------------------------

typedef __bf16 bf16_t;
typedef __attribute__((ext_vector_type(16))) __bf16 v16bf;
typedef __attribute__((ext_vector_type(8)))  __bf16 v8bf;
typedef __attribute__((ext_vector_type(8)))  float  v8f;

#define DIM        512
#define NROWS_TOT  32768
#define NDEPTH     7
#define RROWS      32            // rows per workgroup (2 row tiles of 16)
#define NMAT       5
#define CT_PER_WV  4             // 32 column tiles / 8 waves

// LDS layout (bank-conflict padded)
#define ASTRIDE    520           // bf16 elements per chamber row (512 + 8 pad)
#define OSTRIDE    516           // f32 elements per out-acc row  (512 + 4 pad)
#define A_PART     (RROWS * ASTRIDE)
#define A_BUF      (2 * A_PART)
#define O_PART     (RROWS * OSTRIDE)
#define LDS_BYTES  (2 * A_BUF * 2 + 2 * O_PART * 4)   // 133120 + 132096 = 265216

#define WS_ELEMS   (NMAT * 16 * 32 * 32 * 16)          // 1,310,720 bf16
#define WS_BYTES   (WS_ELEMS * 2)

// ---------------------------------------------------------------------------
// Pass 0: f32 matrices -> bf16 WMMA-B fragment order:
//   ws[(((j*16 + kb)*32 + ct)*32 + lane)*16 + e]
// ---------------------------------------------------------------------------
__global__ __launch_bounds__(256) void convert_mats_kernel(
    const float* __restrict__ m0, const float* __restrict__ m1,
    const float* __restrict__ m2, const float* __restrict__ m3,
    const float* __restrict__ m4, bf16_t* __restrict__ ws)
{
    int flat = blockIdx.x * 256 + threadIdx.x;      // exactly WS_ELEMS threads
    int e    = flat & 15;
    int lane = (flat >> 4)  & 31;
    int ct   = (flat >> 9)  & 31;
    int kb   = (flat >> 14) & 15;
    int j    = flat >> 18;
    int g    = lane >> 4;
    int off  = e + 8 * g + ((e >= 8) ? 8 : 0);
    int K    = 32 * kb + off;
    int N    = 16 * ct + (lane & 15);
    const float* M = (j == 0) ? m0 : (j == 1) ? m1 : (j == 2) ? m2 : (j == 3) ? m3 : m4;
    ws[flat] = (bf16_t)M[K * DIM + N];
}

// ---------------------------------------------------------------------------
// Main fused kernel. 256 threads = 8 waves. grid = 32768/32 = 1024 blocks.
// ---------------------------------------------------------------------------
__global__ __launch_bounds__(256) void resonance_kernel(
    const float*  __restrict__ wre,   const float* __restrict__ wim,
    const bf16_t* __restrict__ wsB,
    const float*  __restrict__ mscale, const float* __restrict__ mbias,
    float* __restrict__ out)
{
    extern __shared__ char smem[];
    bf16_t* Abuf = (bf16_t*)smem;                         // [2 buf][2 part][32][ASTRIDE]
    float*  Osh  = (float*)(smem + 2 * A_BUF * 2);        // [2 part][32][OSTRIDE]

    const int tid     = threadIdx.x;
    const int lane    = tid & 31;
    const int wv      = tid >> 5;
    const int lrow    = lane & 15;
    const int g       = lane >> 4;
    const int rowbase = blockIdx.x * RROWS;

    // --- init: chamber(depth -1) = input wave, out accumulator = 0 ---
    for (int i = tid; i < RROWS * DIM; i += 256) {
        int r = i >> 9, k = i & (DIM - 1);
        Abuf[0 * A_PART + r * ASTRIDE + k] = (bf16_t)wre[(rowbase + r) * DIM + k];
        Abuf[1 * A_PART + r * ASTRIDE + k] = (bf16_t)wim[(rowbase + r) * DIM + k];
        Osh[0 * O_PART + r * OSTRIDE + k] = 0.0f;
        Osh[1 * O_PART + r * OSTRIDE + k] = 0.0f;
    }
    __syncthreads();

    const float wdep[NDEPTH] = {0.3139087f, 0.2249258f, 0.1611665f, 0.1154808f,
                                0.0827455f, 0.0592904f, 0.0424823f};

    #pragma unroll 1
    for (int dep = 0; dep < NDEPTH; ++dep) {
        const bf16_t* Acur  = Abuf + (dep & 1) * A_BUF;
        bf16_t*       Anext = Abuf + ((dep + 1) & 1) * A_BUF;
        const float   wd    = wdep[dep];

        #pragma unroll 1
        for (int cti = 0; cti < CT_PER_WV; ++cti) {
            const int ct   = wv * CT_PER_WV + cti;
            const int dcol = 16 * ct + lrow;

            const v8f vz = {0.f, 0.f, 0.f, 0.f, 0.f, 0.f, 0.f, 0.f};
            v8f Sre[2], Sim[2];          // running harmonic sum (folded)
            v8f c5r[2], c5i[2];          // order-5 GEMM result for scalar fold

            // ======== K-sweep 1: matrices 0 (coupling), 1 (o1), 2 (o2) =====
            {
                v8f cR[2][3], cI[2][3];
                #pragma unroll
                for (int rt = 0; rt < 2; ++rt)
                    #pragma unroll
                    for (int j = 0; j < 3; ++j) { cR[rt][j] = vz; cI[rt][j] = vz; }

                #pragma unroll 4
                for (int kb = 0; kb < 16; ++kb) {
                    v16bf af[2][2];
                    #pragma unroll
                    for (int rt = 0; rt < 2; ++rt)
                        #pragma unroll
                        for (int p = 0; p < 2; ++p) {
                            const bf16_t* base = Acur + p * A_PART +
                                (16 * rt + lrow) * ASTRIDE + 32 * kb + 8 * g;
                            v8bf lo = *(const v8bf*)(base);
                            v8bf hi = *(const v8bf*)(base + 16);
                            af[rt][p] = __builtin_shufflevector(lo, hi,
                                0,1,2,3,4,5,6,7,8,9,10,11,12,13,14,15);
                        }
                    #pragma unroll
                    for (int j = 0; j < 3; ++j) {
                        const bf16_t* bp = wsB + ((((j * 16 + kb) * 32 + ct) * 32) + lane) * 16;
                        v16bf bfrag = *(const v16bf*)bp;
                        #pragma unroll
                        for (int rt = 0; rt < 2; ++rt) {
                            cR[rt][j] = __builtin_amdgcn_wmma_f32_16x16x32_bf16(
                                false, af[rt][0], false, bfrag, (short)0, cR[rt][j], false, false);
                            cI[rt][j] = __builtin_amdgcn_wmma_f32_16x16x32_bf16(
                                false, af[rt][1], false, bfrag, (short)0, cI[rt][j], false, false);
                        }
                    }
                }
                // fold: S = coupling + o1 + 0.25*|o2|^2 (|z|^2 is purely real)
                #pragma unroll
                for (int rt = 0; rt < 2; ++rt) {
                    Sre[rt] = cR[rt][0] + cR[rt][1]
                            + 0.25f * (cR[rt][2] * cR[rt][2] + cI[rt][2] * cI[rt][2]);
                    Sim[rt] = cI[rt][0] + cI[rt][1];
                }
            }

            // ======== K-sweep 2: matrices 3 (o3), 4 (o5) ===================
            {
                v8f cR[2][2], cI[2][2];
                #pragma unroll
                for (int rt = 0; rt < 2; ++rt)
                    #pragma unroll
                    for (int j = 0; j < 2; ++j) { cR[rt][j] = vz; cI[rt][j] = vz; }

                #pragma unroll 4
                for (int kb = 0; kb < 16; ++kb) {
                    v16bf af[2][2];
                    #pragma unroll
                    for (int rt = 0; rt < 2; ++rt)
                        #pragma unroll
                        for (int p = 0; p < 2; ++p) {
                            const bf16_t* base = Acur + p * A_PART +
                                (16 * rt + lrow) * ASTRIDE + 32 * kb + 8 * g;
                            v8bf lo = *(const v8bf*)(base);
                            v8bf hi = *(const v8bf*)(base + 16);
                            af[rt][p] = __builtin_shufflevector(lo, hi,
                                0,1,2,3,4,5,6,7,8,9,10,11,12,13,14,15);
                        }
                    #pragma unroll
                    for (int j = 0; j < 2; ++j) {
                        const bf16_t* bp = wsB + (((((j + 3) * 16 + kb) * 32 + ct) * 32) + lane) * 16;
                        v16bf bfrag = *(const v16bf*)bp;
                        #pragma unroll
                        for (int rt = 0; rt < 2; ++rt) {
                            cR[rt][j] = __builtin_amdgcn_wmma_f32_16x16x32_bf16(
                                false, af[rt][0], false, bfrag, (short)0, cR[rt][j], false, false);
                            cI[rt][j] = __builtin_amdgcn_wmma_f32_16x16x32_bf16(
                                false, af[rt][1], false, bfrag, (short)0, cI[rt][j], false, false);
                        }
                    }
                }
                // fold o3: |z|^2 * z / 9 (vector); keep o5 for scalar fold
                #pragma unroll
                for (int rt = 0; rt < 2; ++rt) {
                    v8f m3 = cR[rt][0] * cR[rt][0] + cI[rt][0] * cI[rt][0];
                    Sre[rt] = Sre[rt] + (1.0f / 9.0f) * (m3 * cR[rt][0]);
                    Sim[rt] = Sim[rt] + (1.0f / 9.0f) * (m3 * cI[rt][0]);
                    c5r[rt] = cR[rt][1];
                    c5i[rt] = cI[rt][1];
                }
            }

            // ======== elementwise: order-5 fold + interference + tanh ======
            const float sc   = mscale[dcol];
            const float bi   = mbias[dcol];
            const float damp = 0.1f / (1.0f + __expf(3.0f * (float)dcol * (1.0f / 511.0f)));
            const float fac  = __expf(-damp * (float)dep);

            #pragma unroll
            for (int rt = 0; rt < 2; ++rt) {
                #pragma unroll
                for (int q = 0; q < 8; ++q) {
                    const int mg = 16 * rt + q + 8 * g;
                    float ire = Sre[rt][q], iim = Sim[rt][q];
                    float x5 = c5r[rt][q], y5 = c5i[rt][q];
                    float r2 = x5 * x5 + y5 * y5;       // mag^(1/5) * e^{i*5*theta}
                    if (r2 > 1e-30f) {
                        float rinv = rsqrtf(r2);
                        float ux = x5 * rinv, uy = y5 * rinv;
                        float u2x = ux * ux - uy * uy,     u2y = 2.f * ux * uy;
                        float u4x = u2x * u2x - u2y * u2y, u4y = 2.f * u2x * u2y;
                        float u5x = u4x * ux - u4y * uy,   u5y = u4x * uy + u4y * ux;
                        float amp = __expf(0.2f * __logf(r2 * rinv));
                        ire += 0.04f * amp * u5x;
                        iim += 0.04f * amp * u5y;
                    }
                    if (dep > 0) {   // interference: superpose previous chamber
                        ire += (float)Acur[0 * A_PART + mg * ASTRIDE + dcol];
                        iim += (float)Acur[1 * A_PART + mg * ASTRIDE + dcol];
                    }
                    float cr = tanhf(ire * sc + bi) * fac;
                    float ci = tanhf(iim * sc + bi) * fac;
                    Anext[0 * A_PART + mg * ASTRIDE + dcol] = (bf16_t)cr;
                    Anext[1 * A_PART + mg * ASTRIDE + dcol] = (bf16_t)ci;
                    Osh[0 * O_PART + mg * OSTRIDE + dcol] += wd * cr;
                    Osh[1 * O_PART + mg * OSTRIDE + dcol] += wd * ci;
                }
            }
        }
        __syncthreads();   // Anext complete before next depth reads it
    }

    // --- write accumulated output [2, B, D] ---
    for (int i = tid; i < RROWS * DIM; i += 256) {
        int r = i >> 9, k = i & (DIM - 1);
        size_t base = (size_t)(rowbase + r) * DIM + k;
        out[base]                           = Osh[0 * O_PART + r * OSTRIDE + k];
        out[(size_t)NROWS_TOT * DIM + base] = Osh[1 * O_PART + r * OSTRIDE + k];
    }
}

// ---------------------------------------------------------------------------
extern "C" void kernel_launch(void* const* d_in, const int* in_sizes, int n_in,
                              void* d_out, int out_size, void* d_ws, size_t ws_size,
                              hipStream_t stream)
{
    (void)in_sizes; (void)n_in; (void)out_size;
    const float* wre  = (const float*)d_in[0];
    const float* wim  = (const float*)d_in[1];
    const float* cmat = (const float*)d_in[2];
    const float* h1   = (const float*)d_in[3];
    const float* h2   = (const float*)d_in[4];
    const float* h3   = (const float*)d_in[5];
    const float* h5   = (const float*)d_in[6];
    const float* msc  = (const float*)d_in[7];
    const float* mbi  = (const float*)d_in[8];

    if (ws_size < (size_t)WS_BYTES) return;   // need 2.5 MB scratch for bf16 matrices
    bf16_t* wsB = (bf16_t*)d_ws;

    convert_mats_kernel<<<WS_ELEMS / 256, 256, 0, stream>>>(cmat, h1, h2, h3, h5, wsB);

    resonance_kernel<<<NROWS_TOT / RROWS, 256, LDS_BYTES, stream>>>(
        wre, wim, wsB, msc, mbi, (float*)d_out);
}
